// LstmModel_27848567948095
// MI455X (gfx1250) — matrix operand order repaired and checked by reading, not compiled, not run on hardware
//
#include <hip/hip_runtime.h>

// ---------------------------------------------------------------------------
// Persistent, layer-pipelined LSTM for MI455X (gfx1250, wave32, WMMA).
//
//   grid = 48 workgroups: 3 layer groups x 16 WGs. WG (layer, wg) owns
//   hidden units [wg*32, wg*32+32) of its layer, i.e. a 128-wide slice of the
//   2048-wide fused gate matmul (4 gates x 32 units).
//
//   Weights (bf16) live in LDS (padded stride, ~258 KB), h-state ring buffers
//   + done-flags + per-step output partials live in d_ws (L2-resident).
//   Matrix core path: v_wmma_f32_16x16x32_bf16, A = h tiles from global
//   (L2-hot, 64 KB per layer), B = W tiles from LDS (row-major along K).
// ---------------------------------------------------------------------------

typedef unsigned short u16;
typedef unsigned int   u32;
typedef __attribute__((ext_vector_type(16))) __bf16 v16bf;
typedef __attribute__((ext_vector_type(8)))  float  v8f;

#define T_STEPS 1024
#define BATCH   64
#define HID     512
#define NWG_L   16          // workgroups per layer
#define UPW     32          // hidden units per WG
#define THREADS 256         // 8 waves (wave32)
#define RING    4           // h-state ring depth (timesteps)

// ---- d_ws layout (bytes) ----
#define DONE_OFF 0                       // 3*1024 ints  (12 KB)
#define HBUF_OFF 16384                   // 3*RING*64*512 bf16 = 768 KB
#define PART_OFF 802816                  // 1024*16*64 f32 = 4 MB

// ---- LDS arena layout (bytes) ----
#define OFF_W   0                        // 128 x 1032 bf16 (padded)  258 KB
#define OFF_G   264192                   // gates   64 x 132 f32      33 KB
#define OFF_C   297984                   // c state 64 x 32 f32        8 KB
#define OFF_HO  306176                   // h out   64 x 33 f32        8.25 KB
#define OFF_B   314624                   // bias 128 f32
#define OFF_WX  315136                   // Wih0 column slice, 128 f32
#define OFF_WO  315648                   // Wout slice, 32 f32
#define LDS_BYTES 315776                 // < 320 KB per-WG LDS limit

__device__ __forceinline__ u16 f2bf(float f) {          // RNE f32 -> bf16 bits
  u32 x = __float_as_uint(f);
  x += 0x7FFFu + ((x >> 16) & 1u);
  return (u16)(x >> 16);
}
__device__ __forceinline__ float sigmf(float x) { return 1.0f / (1.0f + __expf(-x)); }
__device__ __forceinline__ float tanhfast(float x) {
  float xc = fminf(fmaxf(x, -15.0f), 15.0f);
  float e  = __expf(2.0f * xc);
  return (e - 1.0f) / (e + 1.0f);
}

__device__ __forceinline__ void waitCnt(int* p) {
  while (__hip_atomic_load(p, __ATOMIC_RELAXED, __HIP_MEMORY_SCOPE_AGENT) < NWG_L)
    __builtin_amdgcn_s_sleep(1);
}

union FragAB { v16bf v; uint4 q[2]; };

// One branch-free pass of 16 K-steps (K=512): gates[64x128] += h_src x W^T.
//   src    : bf16 h vector [64 x 512] in global memory (L2-hot)
//   wBase  : LDS weight slice, padded row stride WSTR, column base kbBase
__device__ __forceinline__ void mm512(const u16* __restrict__ src,
                                      const u16* __restrict__ wBase,
                                      int WSTR, int kbBase,
                                      int aRow, int kOff, int n4,
                                      int lnLow, int lnHi,
                                      v8f& acc0, v8f& acc1, v8f& acc2, v8f& acc3) {
  const u16* aPtr = src + aRow * HID + kOff;
  const u16* w0 = wBase + (size_t)((n4 + 0) * 16 + lnLow) * WSTR + kbBase + lnHi * 16;
  const u16* w1 = wBase + (size_t)((n4 + 1) * 16 + lnLow) * WSTR + kbBase + lnHi * 16;
  const u16* w2 = wBase + (size_t)((n4 + 2) * 16 + lnLow) * WSTR + kbBase + lnHi * 16;
  const u16* w3 = wBase + (size_t)((n4 + 3) * 16 + lnLow) * WSTR + kbBase + lnHi * 16;
#pragma unroll 4
  for (int ks = 0; ks < 16; ++ks) {
    FragAB A;
    const uint4* gp = (const uint4*)(aPtr + ks * 32);
    A.q[0] = gp[0];        // K = kOff..kOff+7
    A.q[1] = gp[1];        // K = kOff+8..kOff+15   (contiguous, matches B)
    FragAB B0, B1, B2, B3;
    const uint4* bp0 = (const uint4*)(w0 + ks * 32);
    const uint4* bp1 = (const uint4*)(w1 + ks * 32);
    const uint4* bp2 = (const uint4*)(w2 + ks * 32);
    const uint4* bp3 = (const uint4*)(w3 + ks * 32);
    B0.q[0] = bp0[0]; B0.q[1] = bp0[1];
    B1.q[0] = bp1[0]; B1.q[1] = bp1[1];
    B2.q[0] = bp2[0]; B2.q[1] = bp2[1];
    B3.q[0] = bp3[0]; B3.q[1] = bp3[1];
    acc0 = __builtin_amdgcn_wmma_f32_16x16x32_bf16(false, A.v, false, B0.v, (short)0, acc0, false, false);
    acc1 = __builtin_amdgcn_wmma_f32_16x16x32_bf16(false, A.v, false, B1.v, (short)0, acc1, false, false);
    acc2 = __builtin_amdgcn_wmma_f32_16x16x32_bf16(false, A.v, false, B2.v, (short)0, acc2, false, false);
    acc3 = __builtin_amdgcn_wmma_f32_16x16x32_bf16(false, A.v, false, B3.v, (short)0, acc3, false, false);
  }
}

__global__ __launch_bounds__(THREADS, 1)
void lstm_persistent(const float* __restrict__ x,
                     const float* __restrict__ Wih0, const float* __restrict__ Whh0,
                     const float* __restrict__ bih0, const float* __restrict__ bhh0,
                     const float* __restrict__ Wih1, const float* __restrict__ Whh1,
                     const float* __restrict__ bih1, const float* __restrict__ bhh1,
                     const float* __restrict__ Wih2, const float* __restrict__ Whh2,
                     const float* __restrict__ bih2, const float* __restrict__ bhh2,
                     const float* __restrict__ Wout,
                     void* __restrict__ ws) {
  extern __shared__ char smem[];
  u16*   wS  = (u16*)  (smem + OFF_W);
  float* gS  = (float*)(smem + OFF_G);
  float* cS  = (float*)(smem + OFF_C);
  float* hoS = (float*)(smem + OFF_HO);
  float* bS  = (float*)(smem + OFF_B);
  float* wxS = (float*)(smem + OFF_WX);
  float* woS = (float*)(smem + OFF_WO);

  const int tid   = threadIdx.x;
  const int layer = blockIdx.x >> 4;     // 0..2
  const int wg    = blockIdx.x & 15;     // 0..15
  const int u0    = wg * UPW;

  int*   done = (int*)  ((char*)ws + DONE_OFF);
  u16*   hbuf = (u16*)  ((char*)ws + HBUF_OFF);
  float* part = (float*)((char*)ws + PART_OFF);

  const float* Wih = (layer == 0) ? Wih0 : (layer == 1 ? Wih1 : Wih2);
  const float* Whh = (layer == 0) ? Whh0 : (layer == 1 ? Whh1 : Whh2);
  const float* bih = (layer == 0) ? bih0 : (layer == 1 ? bih1 : bih2);
  const float* bhh = (layer == 0) ? bhh0 : (layer == 1 ? bhh1 : bhh2);

  const int WSTR = (layer == 0) ? 520 : 1032;  // padded LDS row stride (bf16 elems)

  // ------------------ setup: stage bf16 weights into LDS ------------------
  // LDS row r (0..127): gate g = r>>5, unit u = u0 + (r&31).
  // layer 0 : cols [0,512)  = Whh0[row][k]
  // layer1/2: cols [0,512)  = Wih[row][k], cols [512,1024) = Whh[row][k-512]
  if (layer == 0) {
    for (int e = tid; e < 128 * 512; e += THREADS) {
      int r = e >> 9, k = e & 511;
      int grow = ((r >> 5) << 9) + u0 + (r & 31);
      wS[r * 520 + k] = f2bf(Whh[grow * 512 + k]);
    }
  } else {
    for (int e = tid; e < 128 * 1024; e += THREADS) {
      int r = e >> 10, k = e & 1023;
      int grow = ((r >> 5) << 9) + u0 + (r & 31);
      float v = (k < 512) ? Wih[grow * 512 + k] : Whh[grow * 512 + (k - 512)];
      wS[r * 1032 + k] = f2bf(v);
    }
  }
  if (tid < 128) {
    int grow = ((tid >> 5) << 9) + u0 + (tid & 31);
    bS[tid]  = bih[grow] + bhh[grow];
    wxS[tid] = (layer == 0) ? Wih0[grow] : 0.0f;   // IO==1: Wih0 is [2048,1]
  }
  if (layer == 2 && tid < UPW) woS[tid] = Wout[u0 + tid];
  for (int e = tid; e < BATCH * UPW; e += THREADS) cS[e] = 0.0f;
  __syncthreads();

  // ------------------ WMMA tiling ------------------
  const int wave  = tid >> 5;
  const int lane  = tid & 31;
  const int lnLow = lane & 15;
  const int lnHi  = lane >> 4;
  const int mTile = wave >> 1;          // 0..3  -> batch rows
  const int n4    = (wave & 1) * 4;     // first of 4 n-tiles -> gate cols
  const int aRow  = mTile * 16 + lnLow; // batch row for A fragment
  const int kOff  = lnHi * 16;          // A-frag K sub-offset (contiguous layout)
  const int kbSelf = (layer == 0) ? 0 : 512;   // recurrent cols in LDS W slice

  for (int s = 0; s < T_STEPS; ++s) {
    // ---- producer/consumer + back-pressure waits (tid 0 spins on L2 flags) ----
    if (tid == 0) {
      if (layer > 0)               waitCnt(&done[(layer - 1) * T_STEPS + s]);       // input ready
      if (s > 0)                   waitCnt(&done[layer * T_STEPS + (s - 1)]);       // self history
      if (s >= RING - 1)           waitCnt(&done[layer * T_STEPS + (s - RING + 1)]);// ring anti-dep (peers)
      if (layer < 2 && s >= RING)  waitCnt(&done[(layer + 1) * T_STEPS + (s - RING)]); // ring anti-dep (next layer)
    }
    __syncthreads();
    __builtin_amdgcn_fence(__ATOMIC_ACQUIRE, "agent");

    const u16* hIn   = hbuf + (size_t)((layer - 1) * RING + (s & 3)) * (BATCH * HID);
    const u16* hSelf = hbuf + (size_t)(layer * RING + ((s - 1) & 3)) * (BATCH * HID);
    u16*       hOut  = hbuf + (size_t)(layer * RING + (s & 3)) * (BATCH * HID);

    // gates[64 x 128] = h_in x Wih^T + h_{s-1} x Whh^T (branch-free passes).
    // At s==0 the recurrent pass is skipped: h_{-1} == 0 contributes nothing.
    v8f acc0 = {}, acc1 = {}, acc2 = {}, acc3 = {};
    if (layer > 0)
      mm512(hIn, wS, WSTR, 0, aRow, kOff, n4, lnLow, lnHi, acc0, acc1, acc2, acc3);
    if (s > 0)
      mm512(hSelf, wS, WSTR, kbSelf, aRow, kOff, n4, lnLow, lnHi, acc0, acc1, acc2, acc3);

    // scatter D tiles to LDS gate buffer (padded stride 132)
#pragma unroll
    for (int v = 0; v < 8; ++v) {
      int gm = mTile * 16 + v + lnHi * 8;       // ISA C/D layout: M = v (+8 hi lanes)
      float* gr = gS + gm * 132 + lnLow;
      gr[(n4 + 0) * 16] = acc0[v];
      gr[(n4 + 1) * 16] = acc1[v];
      gr[(n4 + 2) * 16] = acc2[v];
      gr[(n4 + 3) * 16] = acc3[v];
    }
    __syncthreads();

    // ---- activations + cell update (f32), 8 (b,u) elements per thread ----
    for (int e = tid; e < BATCH * UPW; e += THREADS) {
      int b = e >> 5, u = e & 31;
      float xv = 0.0f;
      if (layer == 0 && s >= 1) xv = x[b * T_STEPS + (s - 1)];   // DELAY=1 shift
      const float* gr = gS + b * 132;
      float gi = gr[u]      + bS[u]      + xv * wxS[u];
      float gf = gr[32 + u] + bS[32 + u] + xv * wxS[32 + u];
      float gg = gr[64 + u] + bS[64 + u] + xv * wxS[64 + u];
      float go = gr[96 + u] + bS[96 + u] + xv * wxS[96 + u];
      float ii = sigmf(gi), ff = sigmf(gf), gt = tanhfast(gg), oo = sigmf(go);
      float c  = ff * cS[e] + ii * gt;
      cS[e] = c;
      float h = oo * tanhfast(c);
      hOut[b * HID + u0 + u] = f2bf(h);
      if (layer == 2) hoS[b * 33 + u] = h;     // padded stride vs bank conflicts
    }

    if (layer == 2) {                          // partial output dot with Wout
      __syncthreads();
      if (tid < BATCH) {
        float sum = 0.0f;
#pragma unroll
        for (int u = 0; u < UPW; ++u) sum += hoS[tid * 33 + u] * woS[u];
        part[((size_t)s * NWG_L + wg) * BATCH + tid] = sum;
      }
    }

    // ---- publish: release h slice, bump done flag ----
    __builtin_amdgcn_fence(__ATOMIC_RELEASE, "agent");
    __syncthreads();
    if (tid == 0)
      __hip_atomic_fetch_add(&done[layer * T_STEPS + s], 1,
                             __ATOMIC_RELAXED, __HIP_MEMORY_SCOPE_AGENT);
  }
}

// zero the cross-WG flags every call (d_ws is not re-poisoned between replays)
__global__ void lstm_init_flags(int* __restrict__ done) {
  int i = blockIdx.x * blockDim.x + threadIdx.x;
  if (i < 3 * T_STEPS) done[i] = 0;
}

// out[b,t] = bout + sum over the 16 layer-2 WG partials
__global__ void lstm_out_reduce(const float* __restrict__ part,
                                const float* __restrict__ bout,
                                float* __restrict__ out) {
  int e = blockIdx.x * blockDim.x + threadIdx.x;   // 65536
  int b = e >> 10, t = e & 1023;
  float sum = bout[0];
#pragma unroll
  for (int p = 0; p < NWG_L; ++p)
    sum += part[((size_t)t * NWG_L + p) * BATCH + b];
  out[b * T_STEPS + t] = sum;
}

extern "C" void kernel_launch(void* const* d_in, const int* in_sizes, int n_in,
                              void* d_out, int out_size, void* d_ws, size_t ws_size,
                              hipStream_t stream) {
  const float* x    = (const float*)d_in[0];
  const float* Wih0 = (const float*)d_in[1];
  const float* Whh0 = (const float*)d_in[2];
  const float* bih0 = (const float*)d_in[3];
  const float* bhh0 = (const float*)d_in[4];
  const float* Wih1 = (const float*)d_in[5];
  const float* Whh1 = (const float*)d_in[6];
  const float* bih1 = (const float*)d_in[7];
  const float* bhh1 = (const float*)d_in[8];
  const float* Wih2 = (const float*)d_in[9];
  const float* Whh2 = (const float*)d_in[10];
  const float* bih2 = (const float*)d_in[11];
  const float* bhh2 = (const float*)d_in[12];
  const float* Wout = (const float*)d_in[13];
  const float* bout = (const float*)d_in[14];
  float* out = (float*)d_out;

  // allow > 64 KB dynamic LDS (no stream work; safe under graph capture)
  hipFuncSetAttribute((const void*)lstm_persistent,
                      hipFuncAttributeMaxDynamicSharedMemorySize, LDS_BYTES);

  int* done = (int*)((char*)d_ws + DONE_OFF);
  lstm_init_flags<<<(3 * T_STEPS + 255) / 256, 256, 0, stream>>>(done);

  // 48 WGs (3 layers x 16) must be co-resident: one WGP each, 8 waves/WGP.
  lstm_persistent<<<48, THREADS, LDS_BYTES, stream>>>(
      x, Wih0, Whh0, bih0, bhh0, Wih1, Whh1, bih1, bhh1,
      Wih2, Whh2, bih2, bhh2, Wout, d_ws);

  const float* part = (const float*)((char*)d_ws + PART_OFF);
  lstm_out_reduce<<<(BATCH * T_STEPS) / 256, 256, 0, stream>>>(part, bout, out);
}
